// GraphAttentionNetwork_62749472195066
// MI455X (gfx1250) — matrix operand order repaired and checked by reading, not compiled, not run on hardware
//
#include <hip/hip_runtime.h>
#include <math.h>

typedef __attribute__((ext_vector_type(16))) _Float16 v16h;
typedef __attribute__((ext_vector_type(8)))  _Float16 v8h;
typedef __attribute__((ext_vector_type(8)))  float    v8f;

#define NN   50000
#define EE   800000
#define ET   850000            // E + N (self edges appended)
#define CIN  144               // 2*64 + 16
#define NTILES 53125           // ET / 16 exactly
#define WQ_FRAG_HALVES (5*4*32*16)   // 10240
#define W1_FRAG_HALVES (2*4*32*16)   // 4096

// ---------------------------------------------------------------------------
// Prep 1: convert nodes / edges to f16 (halves the dominant gather traffic)
// ---------------------------------------------------------------------------
__global__ void prep_cvt_kernel(const float* __restrict__ nodes,
                                const float* __restrict__ edges,
                                _Float16* __restrict__ nodes_h,
                                _Float16* __restrict__ edges_h)
{
    long i = (long)blockIdx.x * blockDim.x + threadIdx.x;
    if (i < (long)NN * 64)  nodes_h[i] = (_Float16)nodes[i];
    if (i < (long)EE * 16)  edges_h[i] = (_Float16)edges[i];
}

// ---------------------------------------------------------------------------
// Prep 2: pack Wq/W0/W1 into per-lane WMMA B-fragment layout (f16).
// Fragment (c,t): lane l holds 16 halves at ((c*4+t)*32 + l)*16, where
// element e maps to k = c*32 + (e&8)*2 + (l>>4)*8 + (e&7), n = t*16 + (l&15).
// ---------------------------------------------------------------------------
__global__ void prep_wfrag_kernel(const float* __restrict__ Wq,
                                  const float* __restrict__ W0,
                                  const float* __restrict__ W1,
                                  _Float16* __restrict__ wq_f,
                                  _Float16* __restrict__ w0_f,
                                  _Float16* __restrict__ w1_f)
{
    int i = blockIdx.x * blockDim.x + threadIdx.x;
    if (i < WQ_FRAG_HALVES) {
        int e = i & 15, lane = (i >> 4) & 31, t = (i >> 9) & 3, c = i >> 11;
        int k = c * 32 + (e & 8) * 2 + (lane >> 4) * 8 + (e & 7);
        int n = t * 16 + (lane & 15);
        wq_f[i] = (_Float16)((k < CIN) ? Wq[k * 64 + n] : 0.f);
        w0_f[i] = (_Float16)((k < CIN) ? W0[k * 64 + n] : 0.f);
    }
    if (i < W1_FRAG_HALVES) {
        int e = i & 15, lane = (i >> 4) & 31, t = (i >> 9) & 3, c = i >> 11;
        int k = c * 32 + (e & 8) * 2 + (lane >> 4) * 8 + (e & 7);
        int n = t * 16 + (lane & 15);
        w1_f[i] = (_Float16)W1[k * 64 + n];
    }
}

// ---------------------------------------------------------------------------
// Kernel 1: per-16-edge tile, WMMA f16 GEMMs. One wave per tile, 4 waves/block.
// LayerNorm done in D-fragment registers via half-wave butterfly reductions.
// ---------------------------------------------------------------------------
__global__ __launch_bounds__(128)
void gat_edge_kernel(const _Float16* __restrict__ nodes_h,
                     const _Float16* __restrict__ edges_h,
                     const int* __restrict__ senders, const int* __restrict__ receivers,
                     const _Float16* __restrict__ wq_f, const float* __restrict__ bq,
                     const _Float16* __restrict__ w0_f, const float* __restrict__ b0,
                     const float* __restrict__ g0, const float* __restrict__ beta0,
                     const _Float16* __restrict__ w1_f, const float* __restrict__ b1,
                     const float* __restrict__ g1, const float* __restrict__ beta1,
                     const float* __restrict__ Wf, const float* __restrict__ bf,
                     _Float16* __restrict__ qout, float* __restrict__ logits)
{
    __shared__ _Float16 hbuf[4][16 * 64];  // per-wave 16x64 f16 normalized tile

    const int lane = threadIdx.x & 31;
    const int wib  = threadIdx.x >> 5;
    const int tile = blockIdx.x * 4 + wib;
    if (tile >= NTILES) return;            // uniform per wave -> EXEC stays all-1s

    _Float16* h16 = hbuf[wib];

    const int row  = lane & 15;
    const int hi   = lane >> 4;
    const int eidx = tile * 16 + row;      // this lane's A-matrix edge row

    int s, r;
    if (eidx < EE) { s = senders[eidx]; r = receivers[eidx]; }
    else           { s = eidx - EE;     r = s; }             // self edge

    const _Float16* __restrict__ srow = nodes_h + (size_t)s * 64;
    const _Float16* __restrict__ rrow = nodes_h + (size_t)r * 64;

    // ---- A fragments: each 8-element K-run is one aligned 16B v8h load.
    // Run starts 32c+8hi / 32c+16+8hi never straddle region boundaries
    // (64 / 128 / 144), so region selection is compile-time after unroll.
    v16h afrag[5];
    {
        const int h8 = hi * 8;
        v8h zero8 = {(_Float16)0, (_Float16)0, (_Float16)0, (_Float16)0,
                     (_Float16)0, (_Float16)0, (_Float16)0, (_Float16)0};
        v8h runs[10];
        runs[0] = *(const v8h*)(srow + 0  + h8);
        runs[1] = *(const v8h*)(srow + 16 + h8);
        runs[2] = *(const v8h*)(srow + 32 + h8);
        runs[3] = *(const v8h*)(srow + 48 + h8);
        runs[4] = *(const v8h*)(rrow + 0  + h8);
        runs[5] = *(const v8h*)(rrow + 16 + h8);
        runs[6] = *(const v8h*)(rrow + 32 + h8);
        runs[7] = *(const v8h*)(rrow + 48 + h8);
        runs[8] = zero8;
        if (eidx < EE) runs[8] = *(const v8h*)(edges_h + (size_t)eidx * 16 + h8);
        runs[9] = zero8;                                   // K pad 144..159
#pragma unroll
        for (int c = 0; c < 5; ++c) {
#pragma unroll
            for (int e = 0; e < 8; ++e) {
                afrag[c][e]     = runs[2 * c][e];
                afrag[c][e + 8] = runs[2 * c + 1][e];
            }
        }
    }

    const v8f vzero = {0.f, 0.f, 0.f, 0.f, 0.f, 0.f, 0.f, 0.f};
    v8f qacc[4], hacc[4];
#pragma unroll
    for (int t = 0; t < 4; ++t) { qacc[t] = vzero; hacc[t] = vzero; }

    // ---- GEMM 1+2 fused over K (padded 160 = 5 chunks); B frags are 32B loads
#pragma unroll
    for (int c = 0; c < 5; ++c) {
#pragma unroll
        for (int t = 0; t < 4; ++t) {
            const size_t fo = (size_t)((c * 4 + t) * 32 + lane) * 16;
            v16h bqf = *(const v16h*)(wq_f + fo);
            v16h b0f = *(const v16h*)(w0_f + fo);
            qacc[t] = __builtin_amdgcn_wmma_f32_16x16x32_f16(false, afrag[c], false, bqf,
                                                             (short)0, qacc[t], false, false);
            hacc[t] = __builtin_amdgcn_wmma_f32_16x16x32_f16(false, afrag[c], false, b0f,
                                                             (short)0, hacc[t], false, false);
        }
    }

    // ---- store q = x*Wq + bq (f16) ; D layout: VGPR rr -> M = rr + hi*8 ----
#pragma unroll
    for (int t = 0; t < 4; ++t) {
        const int n = t * 16 + row;
        const float bias = bq[n];
#pragma unroll
        for (int rr = 0; rr < 8; ++rr) {
            const int m = rr + hi * 8;
            qout[(size_t)(tile * 16 + m) * 64 + n] = (_Float16)(qacc[t][rr] + bias);
        }
    }

    // ---- LN0 in registers: row m = hi*8+rr lives in this half-wave's 16
    // lanes (4 values/lane). Butterfly masks 1..8 stay inside the half.
    {
#pragma unroll
        for (int t = 0; t < 4; ++t) {
            const float bias = b0[t * 16 + row];
#pragma unroll
            for (int rr = 0; rr < 8; ++rr) {
                float v = hacc[t][rr] + bias;
                hacc[t][rr] = v > 0.f ? v : 0.f;
            }
        }
        float s1[8], s2[8];
#pragma unroll
        for (int rr = 0; rr < 8; ++rr) {
            float p = 0.f, p2 = 0.f;
#pragma unroll
            for (int t = 0; t < 4; ++t) { float v = hacc[t][rr]; p += v; p2 += v * v; }
            s1[rr] = p; s2[rr] = p2;
        }
#pragma unroll
        for (int msk = 1; msk <= 8; msk <<= 1) {
#pragma unroll
            for (int rr = 0; rr < 8; ++rr) {
                s1[rr] += __shfl_xor(s1[rr], msk, 32);
                s2[rr] += __shfl_xor(s2[rr], msk, 32);
            }
        }
        float mu[8], inv[8];
#pragma unroll
        for (int rr = 0; rr < 8; ++rr) {
            float m_ = s1[rr] * (1.f / 64.f);
            float v_ = s2[rr] * (1.f / 64.f) - m_ * m_;
            mu[rr] = m_;
            inv[rr] = rsqrtf(v_ + 1e-6f);
        }
        // normalize and write f16 tile to LDS (row-major [16][64]) for GEMM 3
#pragma unroll
        for (int t = 0; t < 4; ++t) {
            const int n = t * 16 + row;
            const float gg = g0[n], bb = beta0[n];
#pragma unroll
            for (int rr = 0; rr < 8; ++rr) {
                float v = (hacc[t][rr] - mu[rr]) * inv[rr] * gg + bb;
                h16[(rr + hi * 8) * 64 + n] = (_Float16)v;
            }
        }
    }

    // ---- GEMM 3: h1 = h0n * W1 (K = 64 -> 2 chunks); A runs are 16B ds loads
    v8f h1acc[4];
#pragma unroll
    for (int t = 0; t < 4; ++t) h1acc[t] = vzero;
#pragma unroll
    for (int c = 0; c < 2; ++c) {
        v16h a;
        {
            const int k0 = c * 32 + hi * 8;
            v8h lo = *(const v8h*)(h16 + row * 64 + k0);
            v8h hh = *(const v8h*)(h16 + row * 64 + k0 + 16);
#pragma unroll
            for (int e = 0; e < 8; ++e) { a[e] = lo[e]; a[e + 8] = hh[e]; }
        }
#pragma unroll
        for (int t = 0; t < 4; ++t) {
            const size_t fo = (size_t)((c * 4 + t) * 32 + lane) * 16;
            v16h bw = *(const v16h*)(w1_f + fo);
            h1acc[t] = __builtin_amdgcn_wmma_f32_16x16x32_f16(false, a, false, bw,
                                                              (short)0, h1acc[t], false, false);
        }
    }

    // ---- LN1 + Dense(1)+relu entirely in registers (no LDS) ----
    {
#pragma unroll
        for (int t = 0; t < 4; ++t) {
            const float bias = b1[t * 16 + row];
#pragma unroll
            for (int rr = 0; rr < 8; ++rr) {
                float v = h1acc[t][rr] + bias;
                h1acc[t][rr] = v > 0.f ? v : 0.f;
            }
        }
        float s1[8], s2[8];
#pragma unroll
        for (int rr = 0; rr < 8; ++rr) {
            float p = 0.f, p2 = 0.f;
#pragma unroll
            for (int t = 0; t < 4; ++t) { float v = h1acc[t][rr]; p += v; p2 += v * v; }
            s1[rr] = p; s2[rr] = p2;
        }
#pragma unroll
        for (int msk = 1; msk <= 8; msk <<= 1) {
#pragma unroll
            for (int rr = 0; rr < 8; ++rr) {
                s1[rr] += __shfl_xor(s1[rr], msk, 32);
                s2[rr] += __shfl_xor(s2[rr], msk, 32);
            }
        }
        float p[8];
#pragma unroll
        for (int rr = 0; rr < 8; ++rr) {
            float m_ = s1[rr] * (1.f / 64.f);
            float v_ = s2[rr] * (1.f / 64.f) - m_ * m_;
            s1[rr] = m_;                      // reuse as mu
            s2[rr] = rsqrtf(v_ + 1e-6f);      // reuse as inv
            p[rr] = 0.f;
        }
#pragma unroll
        for (int t = 0; t < 4; ++t) {
            const int n = t * 16 + row;
            const float gg = g1[n], bb = beta1[n], wf = Wf[n];
#pragma unroll
            for (int rr = 0; rr < 8; ++rr) {
                float v = (h1acc[t][rr] - s1[rr]) * s2[rr] * gg + bb;
                p[rr] += v * wf;
            }
        }
#pragma unroll
        for (int msk = 1; msk <= 8; msk <<= 1) {
#pragma unroll
            for (int rr = 0; rr < 8; ++rr) p[rr] += __shfl_xor(p[rr], msk, 32);
        }
        const float bfv = bf[0];
#pragma unroll
        for (int rr = 0; rr < 8; ++rr) {
            if ((lane & 15) == rr) {
                float l = p[rr] + bfv;
                logits[tile * 16 + hi * 8 + rr] = l > 0.f ? l : 0.f;
            }
        }
    }
}

// ---------------------------------------------------------------------------
__global__ void gat_init_kernel(float* m, float* denom, float* acc)
{
    long i = (long)blockIdx.x * blockDim.x + threadIdx.x;
    if (i < NN) { m[i] = 0.f; denom[i] = 0.f; }
    if (i < (long)NN * 64) acc[i] = 0.f;
}

__global__ void gat_max_kernel(const int* __restrict__ receivers,
                               const float* __restrict__ logits, unsigned* m)
{
    int e = blockIdx.x * blockDim.x + threadIdx.x;
    if (e >= ET) return;
    int r = (e < EE) ? receivers[e] : (e - EE);
    // logits >= 0 (relu) => uint bit order == float order; init 0u == 0.0f valid
    atomicMax(&m[r], __float_as_uint(logits[e]));
}

__global__ void gat_exp_kernel(const int* __restrict__ receivers,
                               const float* __restrict__ logits,
                               const float* __restrict__ m,
                               float* __restrict__ evals, float* denom)
{
    int e = blockIdx.x * blockDim.x + threadIdx.x;
    if (e >= ET) return;
    int r = (e < EE) ? receivers[e] : (e - EE);
    float ev = expf(logits[e] - m[r]);
    evals[e] = ev;
    atomicAdd(&denom[r], ev);
}

__global__ __launch_bounds__(64)
void gat_agg_kernel(const int* __restrict__ receivers,
                    const _Float16* __restrict__ q,
                    const float* __restrict__ evals,
                    const float* __restrict__ denom, float* acc)
{
    int e = blockIdx.x;       // 0 .. ET-1
    int j = threadIdx.x;      // 0 .. 63
    int r = (e < EE) ? receivers[e] : (e - EE);
    float w = evals[e] / denom[r];
    float qv = (float)q[(size_t)e * 64 + j];
    atomicAdd(&acc[(size_t)r * 64 + j], w * qv);
}

__global__ void gat_fin_kernel(const float* __restrict__ acc, float* __restrict__ out)
{
    long i = (long)blockIdx.x * blockDim.x + threadIdx.x;
    if (i >= (long)NN * 64) return;
    float v = acc[i];
    out[i] = v > 0.f ? v : 0.01f * v;   // leaky_relu default slope
}

// ---------------------------------------------------------------------------
extern "C" void kernel_launch(void* const* d_in, const int* in_sizes, int n_in,
                              void* d_out, int out_size, void* d_ws, size_t ws_size,
                              hipStream_t stream)
{
    const float* nodes     = (const float*)d_in[0];
    const float* edges     = (const float*)d_in[1];
    const int*   senders   = (const int*)  d_in[2];
    const int*   receivers = (const int*)  d_in[3];
    const float* Wq    = (const float*)d_in[4];
    const float* bq    = (const float*)d_in[5];
    const float* W0    = (const float*)d_in[6];
    const float* b0    = (const float*)d_in[7];
    const float* g0    = (const float*)d_in[8];
    const float* beta0 = (const float*)d_in[9];
    const float* W1    = (const float*)d_in[10];
    const float* b1    = (const float*)d_in[11];
    const float* g1    = (const float*)d_in[12];
    const float* beta1 = (const float*)d_in[13];
    const float* Wf    = (const float*)d_in[14];
    const float* bf    = (const float*)d_in[15];
    (void)in_sizes; (void)n_in; (void)out_size; (void)ws_size;

    char* ws = (char*)d_ws;
    size_t off = 0;
    auto take = [&](size_t bytes) { size_t o = off; off += (bytes + 255) & ~(size_t)255; return o; };
    _Float16* q       = (_Float16*)(ws + take((size_t)ET * 64 * 2));
    float*    logits  = (float*)   (ws + take((size_t)ET * 4));
    float*    evals   = (float*)   (ws + take((size_t)ET * 4));
    float*    m       = (float*)   (ws + take((size_t)NN * 4));
    float*    denom   = (float*)   (ws + take((size_t)NN * 4));
    float*    acc     = (float*)   (ws + take((size_t)NN * 64 * 4));
    _Float16* nodes_h = (_Float16*)(ws + take((size_t)NN * 64 * 2));
    _Float16* edges_h = (_Float16*)(ws + take((size_t)EE * 16 * 2));
    _Float16* wq_f    = (_Float16*)(ws + take((size_t)WQ_FRAG_HALVES * 2));
    _Float16* w0_f    = (_Float16*)(ws + take((size_t)WQ_FRAG_HALVES * 2));
    _Float16* w1_f    = (_Float16*)(ws + take((size_t)W1_FRAG_HALVES * 2));

    prep_cvt_kernel<<<((long)EE * 16 + 255) / 256, 256, 0, stream>>>(nodes, edges, nodes_h, edges_h);
    prep_wfrag_kernel<<<(WQ_FRAG_HALVES + 255) / 256, 256, 0, stream>>>(Wq, W0, W1, wq_f, w0_f, w1_f);
    gat_init_kernel<<<(NN * 64 + 255) / 256, 256, 0, stream>>>(m, denom, acc);
    gat_edge_kernel<<<(NTILES + 3) / 4, 128, 0, stream>>>(
        nodes_h, edges_h, senders, receivers, wq_f, bq, w0_f, b0, g0, beta0,
        w1_f, b1, g1, beta1, Wf, bf, q, logits);
    gat_max_kernel<<<(ET + 255) / 256, 256, 0, stream>>>(receivers, logits, (unsigned*)m);
    gat_exp_kernel<<<(ET + 255) / 256, 256, 0, stream>>>(receivers, logits, m, evals, denom);
    gat_agg_kernel<<<ET, 64, 0, stream>>>(receivers, q, evals, denom, acc);
    gat_fin_kernel<<<(NN * 64 + 255) / 256, 256, 0, stream>>>(acc, (float*)d_out);
}
